// SparseMoE_22471268892885
// MI455X (gfx1250) — matrix operand order repaired and checked by reading, not compiled, not run on hardware
//
#include <hip/hip_runtime.h>

typedef __attribute__((ext_vector_type(16))) __bf16 v16bf;
typedef __attribute__((ext_vector_type(8)))  float  v8f;

#define BTOK 16384
#define INK  1024
#define NOUT 1024
#define NEXP 8
#define TOPK 2
#define RH   128
#define TEMPR 2.0f

// ---------------------------------------------------------------- utilities
__global__ void zero_meta_kernel(int* meta) {
    if (threadIdx.x < 64) meta[threadIdx.x] = 0;
}

__global__ void zero_f32_kernel(float* p, int n4) {
    int i = blockIdx.x * blockDim.x + threadIdx.x;
    if (i < n4) ((float4*)p)[i] = make_float4(0.f, 0.f, 0.f, 0.f);
}

__global__ void cvt_x_kernel(const float* __restrict__ src, __bf16* __restrict__ dst, int n4) {
    int i = blockIdx.x * blockDim.x + threadIdx.x;
    if (i >= n4) return;
    float4 v = ((const float4*)src)[i];
    __bf16 o[4] = {(__bf16)v.x, (__bf16)v.y, (__bf16)v.z, (__bf16)v.w};
    *(uint2*)(dst + (size_t)i * 4) = *(const uint2*)o;
}

// src: [E][R][C] f32 row-major  ->  dst: [E][C][R] bf16 (K-contiguous per N row)
__global__ void cvt_wt_kernel(const float* __restrict__ src, __bf16* __restrict__ dst,
                              int R, int C, long total) {
    long idx = (long)blockIdx.x * blockDim.x + threadIdx.x;
    if (idx >= total) return;
    long rc = (long)R * C;
    int  e  = (int)(idx / rc);
    long rem = idx - (long)e * rc;
    int  r = (int)(rem / C);
    int  c = (int)(rem - (long)r * C);
    dst[(size_t)e * rc + (size_t)c * R + r] = (__bf16)src[idx];
}

// ---------------------------------------------------------------- router
__device__ inline unsigned hmix(unsigned x) {
    x ^= x >> 16; x *= 0x7feb352dU;
    x ^= x >> 15; x *= 0x846ca68bU;
    x ^= x >> 16; return x;
}

__global__ __launch_bounds__(128)
void router_kernel(const float* __restrict__ x,
                   const float* __restrict__ rw1, const float* __restrict__ rb1,
                   const float* __restrict__ rw2, const float* __restrict__ rb2,
                   const float* __restrict__ nw,  const float* __restrict__ nb,
                   float* __restrict__ routing_out,
                   int* __restrict__ topk, float* __restrict__ gates,
                   int* __restrict__ counts) {
    __shared__ float xs[INK];
    __shared__ float hs[RH];
    __shared__ float zs[NEXP];
    const int b = blockIdx.x;
    const int t = threadIdx.x;
    const float* xr = x + (size_t)b * INK;
    for (int j = 0; j < INK / RH; ++j) xs[t + RH * j] = xr[t + RH * j];
    __syncthreads();

    // hidden: h[t] = relu(x . rw1[:,t] + rb1[t])
    float acc = rb1[t];
    #pragma unroll 4
    for (int i = 0; i < INK; ++i) acc += xs[i] * rw1[i * RH + t];
    hs[t] = fmaxf(acc, 0.f);
    __syncthreads();

    if (t < NEXP) {
        const int e = t;
        float lg = rb2[e];
        for (int j = 0; j < RH; ++j) lg += hs[j] * rw2[j * NEXP + e];
        float na = nb[e];
        #pragma unroll 4
        for (int i = 0; i < INK; ++i) na += xs[i] * nw[i * NEXP + e];
        float nscale = (na > 20.f) ? na : log1pf(expf(na));
        // deterministic pseudo-normal (stands in for threefry normal; compile-only)
        unsigned s  = (unsigned)(b * NEXP + e);
        unsigned u1 = hmix(s ^ 0x9E3779B9u);
        unsigned u2 = hmix(s + 0x85EBCA6Bu);
        float f1 = ((float)u1 + 1.0f) * (1.0f / 4294967296.0f);
        float f2 = (float)u2 * (1.0f / 4294967296.0f);
        float eps = sqrtf(-2.f * logf(f1)) * cosf(6.28318530718f * f2);
        zs[e] = (lg + eps * nscale) / TEMPR;
    }
    __syncthreads();

    if (t == 0) {
        float m = zs[0];
        for (int e = 1; e < NEXP; ++e) m = fmaxf(m, zs[e]);
        float p[NEXP], sum = 0.f;
        for (int e = 0; e < NEXP; ++e) { p[e] = expf(zs[e] - m); sum += p[e]; }
        float inv = 1.f / sum;
        for (int e = 0; e < NEXP; ++e) {
            p[e] *= inv;
            routing_out[(size_t)b * NEXP + e] = p[e];
        }
        int i0 = 0;
        for (int e = 1; e < NEXP; ++e) if (p[e] > p[i0]) i0 = e;
        int i1 = (i0 == 0) ? 1 : 0;
        for (int e = 0; e < NEXP; ++e)
            if (e != i0 && p[e] > p[i1]) i1 = e;
        topk[b * TOPK + 0] = i0;  gates[b * TOPK + 0] = p[i0];
        topk[b * TOPK + 1] = i1;  gates[b * TOPK + 1] = p[i1];
        atomicAdd(&counts[i0], 1);
        atomicAdd(&counts[i1], 1);
    }
}

__global__ void offsets_kernel(int* meta) {
    // meta[0..7]=counts, meta[8..16]=offsets, meta[20..27]=cursors
    if (threadIdx.x == 0 && blockIdx.x == 0) {
        int off = 0;
        for (int e = 0; e < NEXP; ++e) {
            meta[8 + e]  = off;
            meta[20 + e] = off;
            off += meta[e];
        }
        meta[8 + NEXP] = off;
    }
}

__global__ void scatter_kernel(const int* __restrict__ topk, const float* __restrict__ gates,
                               int* __restrict__ cursors,
                               int* __restrict__ tokList, float* __restrict__ gateList) {
    int b = blockIdx.x * blockDim.x + threadIdx.x;
    if (b >= BTOK) return;
    #pragma unroll
    for (int k = 0; k < TOPK; ++k) {
        int e = topk[b * TOPK + k];
        int pos = atomicAdd(&cursors[e], 1);
        tokList[pos]  = b;
        gateList[pos] = gates[b * TOPK + k];
    }
}

// ---------------------------------------------------------------- grouped GEMM
#define LDS_STRIDE 40   // 32 live bf16 + 8 pad => 80B row, conflict-spreading
#define NKSTEP (INK / 32)

__device__ __forceinline__ void async_b128(unsigned lds_addr, unsigned long long gaddr) {
    // CDNA5 async DMA: LDS[vdst] = MEM[vaddr], 16B/lane, tracked by ASYNCcnt
    asm volatile("global_load_async_to_lds_b128 %0, %1, off"
                 :: "v"(lds_addr), "v"(gaddr) : "memory");
}
__device__ __forceinline__ unsigned lds_off(const void* p) {
    return (unsigned)(unsigned long long)(uintptr_t)p;   // low 32 bits = LDS byte addr
}

// LAYER 1: A = x_bf16 gathered by token list, out = relu(A*W1t^T + b1) -> h (bf16)
// LAYER 2: A = h (compact rows),             out = gate * (A*W2t^T + b2) atom-> fused
template <int LAYER>
__global__ __launch_bounds__(256)
void moe_gemm_kernel(const int* __restrict__ meta,
                     const int* __restrict__ tokList, const float* __restrict__ gateList,
                     const __bf16* __restrict__ Aglob, const __bf16* __restrict__ Wt,
                     const float* __restrict__ bias,
                     __bf16* __restrict__ hb, float* __restrict__ fused) {
    const int e      = blockIdx.z;
    const int count  = meta[e];           // counts
    const int mBase  = blockIdx.x * 64;
    if (mBase >= count) return;
    const int expOff = meta[8 + e];       // offsets
    const int nBase  = blockIdx.y * 256;

    __shared__ __align__(16) __bf16 As[2][64][LDS_STRIDE];
    __shared__ __align__(16) __bf16 Bs[2][256][LDS_STRIDE];

    const int tid  = threadIdx.x;
    const int lane = tid & 31;
    const int wid  = tid >> 5;
    const int wy   = wid >> 2;            // 0..1 : 32 rows each
    const int wx   = wid & 3;             // 0..3 : 64 cols each
    const int nl   = lane & 15;
    const int kb   = (lane < 16) ? 0 : 8; // ISA 16-bit A/B lane K-base

    v8f acc[2][4];
    #pragma unroll
    for (int a = 0; a < 2; ++a)
        #pragma unroll
        for (int bb = 0; bb < 4; ++bb)
            acc[a][bb] = (v8f)(0.f);

    // A-load coords: 4 threads per row, 8 bf16 (16B) each
    const int ar = tid >> 2;
    const int ac = (tid & 3) * 8;
    const __bf16* aRowPtr;
    if (LAYER == 1) {
        int lr = mBase + ar;
        int cl = (lr < count) ? lr : (count - 1);
        int tok = tokList[expOff + cl];
        aRowPtr = Aglob + (size_t)tok * INK;
    } else {
        aRowPtr = Aglob + ((size_t)expOff + mBase + ar) * INK;  // padded rows in-bounds
    }
    const __bf16* bRowPtr = Wt + ((size_t)e * NOUT + nBase + tid) * INK;

    // per-stage LDS destinations (constant across K loop)
    unsigned ldsA0 = lds_off(&As[0][ar][ac]),  ldsA1 = lds_off(&As[1][ar][ac]);
    unsigned ldsB0 = lds_off(&Bs[0][tid][0]),  ldsB1 = lds_off(&Bs[1][tid][0]);

    // prologue: stage 0 -> buffer 0
    async_b128(ldsA0,      (unsigned long long)(aRowPtr + ac));
    async_b128(ldsB0,      (unsigned long long)(bRowPtr));
    async_b128(ldsB0 + 16, (unsigned long long)(bRowPtr + 8));
    async_b128(ldsB0 + 32, (unsigned long long)(bRowPtr + 16));
    async_b128(ldsB0 + 48, (unsigned long long)(bRowPtr + 24));

    #pragma unroll 2
    for (int kk = 0; kk < NKSTEP; ++kk) {
        const int cur = kk & 1;
        if (kk + 1 < NKSTEP) {
            const int k1 = (kk + 1) * 32;
            async_b128(cur ? ldsA0 : ldsA1, (unsigned long long)(aRowPtr + k1 + ac));
            unsigned bdst = cur ? ldsB0 : ldsB1;
            async_b128(bdst,      (unsigned long long)(bRowPtr + k1));
            async_b128(bdst + 16, (unsigned long long)(bRowPtr + k1 + 8));
            async_b128(bdst + 32, (unsigned long long)(bRowPtr + k1 + 16));
            async_b128(bdst + 48, (unsigned long long)(bRowPtr + k1 + 24));
            asm volatile("s_wait_asynccnt 0x5" ::: "memory");  // stage kk complete
        } else {
            asm volatile("s_wait_asynccnt 0x0" ::: "memory");
        }
        __syncthreads();

        v16bf af[2], bf_[4];
        #pragma unroll
        for (int a = 0; a < 2; ++a) {
            const __bf16* row = &As[cur][wy * 32 + a * 16 + nl][0];
            union { v16bf v; uint4 u[2]; } t;
            t.u[0] = *(const uint4*)(row + kb);
            t.u[1] = *(const uint4*)(row + kb + 16);
            af[a] = t.v;
        }
        #pragma unroll
        for (int bb = 0; bb < 4; ++bb) {
            const __bf16* row = &Bs[cur][wx * 64 + bb * 16 + nl][0];
            union { v16bf v; uint4 u[2]; } t;
            t.u[0] = *(const uint4*)(row + kb);
            t.u[1] = *(const uint4*)(row + kb + 16);
            bf_[bb] = t.v;
        }

        #pragma unroll
        for (int a = 0; a < 2; ++a)
            #pragma unroll
            for (int bb = 0; bb < 4; ++bb)
                acc[a][bb] = __builtin_amdgcn_wmma_f32_16x16x32_bf16(
                    false, af[a], false, bf_[bb], (short)0, acc[a][bb], false, false);
        __syncthreads();   // all reads of buffer `cur` done before it is re-filled
    }

    // epilogue: 16x16 f32 C layout -> (m = v + (lane>=16?8:0), n = lane&15)
    const int rowHalf = (lane >= 16) ? 8 : 0;
    #pragma unroll
    for (int a = 0; a < 2; ++a) {
        #pragma unroll
        for (int bb = 0; bb < 4; ++bb) {
            const int col = nBase + wx * 64 + bb * 16 + nl;
            const float bv = bias[(size_t)e * NOUT + col];
            #pragma unroll
            for (int v = 0; v < 8; ++v) {
                int lr = mBase + wy * 32 + a * 16 + rowHalf + v;   // local row in expert
                if (lr >= count) continue;
                float val = acc[a][bb][v] + bv;
                if (LAYER == 1) {
                    val = fmaxf(val, 0.f);
                    hb[((size_t)expOff + lr) * NOUT + col] = (__bf16)val;
                } else {
                    int gRow = expOff + lr;
                    int tok  = tokList[gRow];
                    float g  = gateList[gRow];
                    atomicAdd(fused + (size_t)tok * NOUT + col, g * val);
                }
            }
        }
    }
}

// ---------------------------------------------------------------- launch
extern "C" void kernel_launch(void* const* d_in, const int* in_sizes, int n_in,
                              void* d_out, int out_size, void* d_ws, size_t ws_size,
                              hipStream_t stream) {
    (void)in_sizes; (void)n_in; (void)out_size; (void)ws_size;
    const float* x   = (const float*)d_in[0];
    const float* rw1 = (const float*)d_in[1];
    const float* rb1 = (const float*)d_in[2];
    const float* rw2 = (const float*)d_in[3];
    const float* rb2 = (const float*)d_in[4];
    const float* nw  = (const float*)d_in[5];
    const float* nb  = (const float*)d_in[6];
    const float* ew1 = (const float*)d_in[7];
    const float* eb1 = (const float*)d_in[8];
    const float* ew2 = (const float*)d_in[9];
    const float* eb2 = (const float*)d_in[10];

    float* fused   = (float*)d_out;
    float* routing = fused + (size_t)BTOK * NOUT;

    // workspace carve-up (all chunks 256B-aligned)
    char* w = (char*)d_ws;
    size_t off = 0;
    int*    meta     = (int*)(w + off);          off += 256;
    int*    topk     = (int*)(w + off);          off += (size_t)BTOK * TOPK * 4;
    float*  gates    = (float*)(w + off);        off += (size_t)BTOK * TOPK * 4;
    int*    tokList  = (int*)(w + off);          off += (size_t)BTOK * TOPK * 4;
    float*  gateList = (float*)(w + off);        off += (size_t)BTOK * TOPK * 4;
    __bf16* xb       = (__bf16*)(w + off);       off += (size_t)BTOK * INK * 2;
    __bf16* w1t      = (__bf16*)(w + off);       off += (size_t)NEXP * INK * NOUT * 2;
    __bf16* w2t      = (__bf16*)(w + off);       off += (size_t)NEXP * NOUT * NOUT * 2;
    __bf16* hb       = (__bf16*)(w + off);       off += ((size_t)BTOK * TOPK + 128) * NOUT * 2;

    // 1. zero meta
    zero_meta_kernel<<<1, 64, 0, stream>>>(meta);

    // 2. convert x -> bf16
    {
        int n4 = BTOK * INK / 4;
        cvt_x_kernel<<<(n4 + 255) / 256, 256, 0, stream>>>(x, xb, n4);
    }
    // 3. transpose+convert weights -> [E][N][K] bf16
    {
        long tot = (long)NEXP * INK * NOUT;
        int blocks = (int)((tot + 255) / 256);
        cvt_wt_kernel<<<blocks, 256, 0, stream>>>(ew1, w1t, INK, NOUT, tot);
        cvt_wt_kernel<<<blocks, 256, 0, stream>>>(ew2, w2t, NOUT, NOUT, tot);
    }

    // 4. router + top-2 + counts
    router_kernel<<<BTOK, RH, 0, stream>>>(x, rw1, rb1, rw2, rb2, nw, nb,
                                           routing, topk, gates, meta /*counts*/);
    // 5. prefix-sum offsets + cursors
    offsets_kernel<<<1, 32, 0, stream>>>(meta);
    // 6. scatter assignment lists
    scatter_kernel<<<(BTOK + 255) / 256, 256, 0, stream>>>(topk, gates, meta + 20,
                                                           tokList, gateList);
    // 7. zero fused output
    {
        int n4 = BTOK * NOUT / 4;
        zero_f32_kernel<<<(n4 + 255) / 256, 256, 0, stream>>>(fused, n4);
    }

    // 8. layer-1 grouped GEMM: h = relu(x W1 + b1)  (bf16 out)
    dim3 grid(BTOK / 64, NOUT / 256, NEXP);   // worst-case tiles; guarded by count
    moe_gemm_kernel<1><<<grid, 256, 0, stream>>>(meta, tokList, gateList,
                                                 xb, w1t, eb1, hb, fused);
    // 9. layer-2 grouped GEMM: fused += gate * (h W2 + b2)
    moe_gemm_kernel<2><<<grid, 256, 0, stream>>>(meta, tokList, gateList,
                                                 hb, w2t, eb2, hb, fused);
}
